// MoELayer_46102178955626
// MI455X (gfx1250) — matrix operand order repaired and checked by reading, not compiled, not run on hardware
//
#include <hip/hip_runtime.h>
#include <hip/hip_bf16.h>

// ---------------- problem constants ----------------
#define T_TOK   8192     // B*S
#define DMODEL  1024
#define HIDDEN  2816
#define NEXP    8

typedef __bf16 bf16;
typedef __attribute__((ext_vector_type(8)))  __bf16 bf16x8;
typedef __attribute__((ext_vector_type(16))) __bf16 bf16x16;
typedef __attribute__((ext_vector_type(8)))  float  f32x8;

// ---------------- workspace layout (bytes, 256-aligned) ----------------
static constexpr size_t OFF_XBF = 0;            // T*D bf16            = 16,777,216
static constexpr size_t OFF_HID = 16777216;     // 25600 rows * H bf16 = 144,179,200
static constexpr size_t OFF_CTR = 160956416;    // 3*T*D f32           = 100,663,296
static constexpr size_t OFF_TW  = 261619712;    // T*2 f32
static constexpr size_t OFF_LT  = 261685248;    // 8*T int
static constexpr size_t OFF_LK  = 261947392;    // 8*T int
static constexpr size_t OFF_CNT = 262209536;    // counts (64B)
static constexpr size_t OFF_OFS = 262209600;    // offsets (64B)

// ---------------- small helpers ----------------
__device__ inline bf16x16 cat16(bf16x8 lo, bf16x8 hi) {
  return __builtin_shufflevector(lo, hi, 0,1,2,3,4,5,6,7,8,9,10,11,12,13,14,15);
}
// A fragment (16x32 MxK bf16): lane holds row M=lane&15; two contiguous 8-K runs
// at k = (lane>>4)*8 and 16+(lane>>4)*8 (ISA 7.12.2 16-bit A layout).
__device__ inline bf16x16 ldsA(const bf16* row, int khalf) {
  bf16x8 lo = *(const bf16x8*)(row + khalf * 8);
  bf16x8 hi = *(const bf16x8*)(row + 16 + khalf * 8);
  return cat16(lo, hi);
}
// B fragment (32x16 KxN bf16): lane holds col N=lane&15; two contiguous 8-K runs
// at k = (lane>>4)*8 and 16+(lane>>4)*8 (same order as A).
__device__ inline bf16x16 ldsB(const bf16* row, int khalf) {
  bf16x8 lo = *(const bf16x8*)(row + khalf * 8);
  bf16x8 hi = *(const bf16x8*)(row + 16 + khalf * 8);
  return cat16(lo, hi);
}
__device__ inline f32x8 bwmma(bf16x16 a, bf16x16 b, f32x8 c) {
  return __builtin_amdgcn_wmma_f32_16x16x32_bf16(false, a, false, b, (short)0, c,
                                                 false, false);
}
__device__ inline bf16x8 cvt8(float4 a, float4 b) {
  bf16x8 o;
  o[0] = (bf16)a.x; o[1] = (bf16)a.y; o[2] = (bf16)a.z; o[3] = (bf16)a.w;
  o[4] = (bf16)b.x; o[5] = (bf16)b.y; o[6] = (bf16)b.z; o[7] = (bf16)b.w;
  return o;
}
__device__ inline unsigned pack_bf16(float lo, float hi) {
  unsigned l = (unsigned)__builtin_bit_cast(unsigned short, (bf16)lo);
  unsigned h = (unsigned)__builtin_bit_cast(unsigned short, (bf16)hi);
  return l | (h << 16);
}
// silu(g)*u with v_rcp_f32 instead of the IEEE divide expansion (result is
// rounded to bf16 anyway; rcp+exp are TRANS ops that co-issue with WMMA).
__device__ inline float silu_mul(float g, float u) {
  return g * __builtin_amdgcn_rcpf(1.f + __expf(-g)) * u;
}

// ---------------- phase 0: x fp32 -> bf16 ----------------
__global__ __launch_bounds__(256) void convert_x_kernel(const float* __restrict__ x,
                                                        bf16* __restrict__ xbf) {
  size_t i = ((size_t)blockIdx.x * 256 + threadIdx.x) * 8;
  float4 a = *(const float4*)(x + i);
  float4 b = *(const float4*)(x + i + 4);
  *(bf16x8*)(xbf + i) = cvt8(a, b);
}

// ---------------- phase 1: router (one wave per token) ----------------
__global__ __launch_bounds__(256) void router_kernel(
    const float* __restrict__ x, const float* __restrict__ rw,
    const float* __restrict__ ebias, float* __restrict__ tokw,
    int* __restrict__ ltok, int* __restrict__ lk, unsigned* __restrict__ counts) {
  int wave = threadIdx.x >> 5, lane = threadIdx.x & 31;
  int t = blockIdx.x * 8 + wave;
  const float* xr = x + (size_t)t * DMODEL;
  float acc[NEXP];
#pragma unroll
  for (int e = 0; e < NEXP; ++e) acc[e] = 0.f;
  for (int i = 0; i < DMODEL / 32; ++i) {
    int dd = i * 32 + lane;
    float xv = xr[dd];
#pragma unroll
    for (int e = 0; e < NEXP; ++e) acc[e] += xv * rw[e * DMODEL + dd];
  }
#pragma unroll
  for (int e = 0; e < NEXP; ++e)
    for (int off = 16; off; off >>= 1) acc[e] += __shfl_xor(acc[e], off, 32);
  if (lane == 0) {
    float s[NEXP];
#pragma unroll
    for (int e = 0; e < NEXP; ++e)
      s[e] = __builtin_amdgcn_rcpf(1.f + __expf(-(acc[e] + ebias[e])));
    int e1 = 0; float s1 = s[0];
#pragma unroll
    for (int e = 1; e < NEXP; ++e) if (s[e] > s1) { s1 = s[e]; e1 = e; }
    int e2 = -1; float s2 = -1e30f;
#pragma unroll
    for (int e = 0; e < NEXP; ++e)
      if (e != e1 && s[e] > s2) { s2 = s[e]; e2 = e; }
    float rd = __builtin_amdgcn_rcpf(s1 + s2 + 1e-6f);
    tokw[2 * t + 0] = s1 * rd;
    tokw[2 * t + 1] = s2 * rd;
    unsigned i1 = atomicAdd(&counts[e1], 1u);
    ltok[e1 * T_TOK + i1] = t; lk[e1 * T_TOK + i1] = 0;
    unsigned i2 = atomicAdd(&counts[e2], 1u);
    ltok[e2 * T_TOK + i2] = t; lk[e2 * T_TOK + i2] = 1;
  }
}

// ---------------- phase 2: 128-padded exclusive scan of counts ----------------
__global__ void scan_kernel(const unsigned* __restrict__ counts,
                            unsigned* __restrict__ offs) {
  if (threadIdx.x == 0 && blockIdx.x == 0) {
    unsigned off = 0;
    for (int e = 0; e < NEXP; ++e) { offs[e] = off; off += (counts[e] + 127u) & ~127u; }
    offs[NEXP] = off;  // shared-expert region (exactly T_TOK rows)
  }
}

// ---------------- phase 3: gate/up GEMMs + SwiGLU -> hidden bf16 ----------------
// grid (64 token-blocks, 22 h-blocks, 9 experts[8=shared]); 512 thr = 16 waves.
// D = W_tile(A, 16x32) x X_tile(B, 32x16): C lanes = tokens, C elements = h.
// Wave w: token-tiles {2*(w&3),+1} (B side), h-tiles {2*(w>>2),+1} (A side);
// per K-step: 4 weight A-frags + 2 token B-frags -> 8 WMMAs.
__global__ __launch_bounds__(512) void gateup_kernel(
    const bf16* __restrict__ xbf,
    const float* __restrict__ gate_w, const float* __restrict__ up_w,
    const float* __restrict__ sgate_w, const float* __restrict__ sup_w,
    const int* __restrict__ ltok,
    const unsigned* __restrict__ counts, const unsigned* __restrict__ offs,
    bf16* __restrict__ hidden) {
  int e = blockIdx.z, hb = blockIdx.y, mb = blockIdx.x;
  unsigned n_e = (e < NEXP) ? counts[e] : (unsigned)T_TOK;
  if ((unsigned)(mb * 128) >= n_e) return;
  const float* gW = (e < NEXP) ? gate_w + (size_t)e * HIDDEN * DMODEL : sgate_w;
  const float* uW = (e < NEXP) ? up_w + (size_t)e * HIDDEN * DMODEL : sup_w;
  unsigned rowBase = offs[e] + mb * 128;

  __shared__ bf16 Xs[128][40];   // token rows x K
  __shared__ bf16 Bg[128][40];   // gate rows (h) x K
  __shared__ bf16 Bu[128][40];   // up rows (h) x K

  int tid = threadIdx.x;
  int r = tid >> 2, seg = tid & 3;
  int idx = mb * 128 + r;
  int tok;
  if (e == NEXP) tok = idx;
  else tok = ((unsigned)idx < n_e) ? ltok[e * T_TOK + idx]
                                   : ltok[e * T_TOK + (int)n_e - 1];
  const bf16* xrow = xbf + (size_t)tok * DMODEL + seg * 8;
  int h = hb * 128 + r;
  const float* grow = gW + (size_t)h * DMODEL + seg * 8;
  const float* urow = uW + (size_t)h * DMODEL + seg * 8;

  int wave = tid >> 5, lane = tid & 31;
  int wTg = wave & 3, wHg = wave >> 2;       // token-group / h-group
  int khalf = lane >> 4, nlo = lane & 15;
  int tRow0 = (2 * wTg) * 16 + nlo, tRow1 = tRow0 + 16;   // B-frag rows (tokens)
  int hRow0 = (2 * wHg) * 16 + nlo, hRow1 = hRow0 + 16;   // A-frag rows (h)

  f32x8 accG[2][2] = {};   // [h-tile][token-tile]
  f32x8 accU[2][2] = {};

  for (int kk = 0; kk < DMODEL; kk += 32) {
    bf16x8 xa = *(const bf16x8*)(xrow + kk);
    float4 g0 = *(const float4*)(grow + kk);
    float4 g1 = *(const float4*)(grow + kk + 4);
    float4 u0 = *(const float4*)(urow + kk);
    float4 u1 = *(const float4*)(urow + kk + 4);
    __syncthreads();
    *(bf16x8*)(&Xs[r][seg * 8]) = xa;
    *(bf16x8*)(&Bg[r][seg * 8]) = cvt8(g0, g1);
    *(bf16x8*)(&Bu[r][seg * 8]) = cvt8(u0, u1);
    __syncthreads();
    bf16x16 ag0 = ldsA(&Bg[hRow0][0], khalf);
    bf16x16 ag1 = ldsA(&Bg[hRow1][0], khalf);
    bf16x16 au0 = ldsA(&Bu[hRow0][0], khalf);
    bf16x16 au1 = ldsA(&Bu[hRow1][0], khalf);
    bf16x16 bx0 = ldsB(&Xs[tRow0][0], khalf);
    bf16x16 bx1 = ldsB(&Xs[tRow1][0], khalf);
    accG[0][0] = bwmma(ag0, bx0, accG[0][0]);
    accG[0][1] = bwmma(ag0, bx1, accG[0][1]);
    accG[1][0] = bwmma(ag1, bx0, accG[1][0]);
    accG[1][1] = bwmma(ag1, bx1, accG[1][1]);
    accU[0][0] = bwmma(au0, bx0, accU[0][0]);
    accU[0][1] = bwmma(au0, bx1, accU[0][1]);
    accU[1][0] = bwmma(au1, bx0, accU[1][0]);
    accU[1][1] = bwmma(au1, bx1, accU[1][1]);
  }

  // Epilogue: lane = token, elements = consecutive h -> pack dwords in-lane.
  int hOff = khalf * 8;
#pragma unroll
  for (int ti = 0; ti < 2; ++ti) {
    int row = rowBase + (2 * wTg + ti) * 16 + nlo;  // padding rows harmless
    size_t rr = (size_t)row * HIDDEN + hb * 128;
#pragma unroll
    for (int hi = 0; hi < 2; ++hi) {
      size_t cb = rr + (2 * wHg + hi) * 16 + hOff;
#pragma unroll
      for (int el = 0; el < 8; el += 2) {
        float h0 = silu_mul(accG[hi][ti][el],     accU[hi][ti][el]);
        float h1 = silu_mul(accG[hi][ti][el + 1], accU[hi][ti][el + 1]);
        *reinterpret_cast<unsigned*>(&hidden[cb + el]) = pack_bf16(h0, h1);
      }
    }
  }
}

// ---------------- phase 4: down GEMM -> weighted contribution planes ----------------
// grid (64 token-blocks, 8 d-blocks, 9 experts); 512 thr; K=2816.
// D = Wd_tile(A) x H_tile(B): C lanes = tokens, C elements = d.
__global__ __launch_bounds__(512) void down_kernel(
    const bf16* __restrict__ hidden,
    const float* __restrict__ down_w, const float* __restrict__ sdown_w,
    const int* __restrict__ ltok, const int* __restrict__ lk,
    const float* __restrict__ tokw,
    const unsigned* __restrict__ counts, const unsigned* __restrict__ offs,
    float* __restrict__ contrib) {
  int e = blockIdx.z, db = blockIdx.y, mb = blockIdx.x;
  unsigned n_e = (e < NEXP) ? counts[e] : (unsigned)T_TOK;
  if ((unsigned)(mb * 128) >= n_e) return;
  const float* dW = (e < NEXP) ? down_w + (size_t)e * DMODEL * HIDDEN : sdown_w;
  unsigned rowBase = offs[e] + mb * 128;

  __shared__ bf16 Ah[128][40];   // token rows x K
  __shared__ bf16 Bd[128][40];   // d rows x K

  int tid = threadIdx.x;
  int r = tid >> 2, seg = tid & 3;
  const bf16* arow = hidden + (size_t)(rowBase + r) * HIDDEN + seg * 8;
  int d = db * 128 + r;
  const float* drow = dW + (size_t)d * HIDDEN + seg * 8;

  int wave = tid >> 5, lane = tid & 31;
  int wTg = wave & 3, wDg = wave >> 2;
  int khalf = lane >> 4, nlo = lane & 15;
  int tRow0 = (2 * wTg) * 16 + nlo, tRow1 = tRow0 + 16;
  int dRow0 = (2 * wDg) * 16 + nlo, dRow1 = dRow0 + 16;

  f32x8 acc[2][2] = {};   // [d-tile][token-tile]

  for (int kk = 0; kk < HIDDEN; kk += 32) {
    bf16x8 av = *(const bf16x8*)(arow + kk);
    float4 d0 = *(const float4*)(drow + kk);
    float4 d1 = *(const float4*)(drow + kk + 4);
    __syncthreads();
    *(bf16x8*)(&Ah[r][seg * 8]) = av;
    *(bf16x8*)(&Bd[r][seg * 8]) = cvt8(d0, d1);
    __syncthreads();
    bf16x16 ad0 = ldsA(&Bd[dRow0][0], khalf);
    bf16x16 ad1 = ldsA(&Bd[dRow1][0], khalf);
    bf16x16 bh0 = ldsB(&Ah[tRow0][0], khalf);
    bf16x16 bh1 = ldsB(&Ah[tRow1][0], khalf);
    acc[0][0] = bwmma(ad0, bh0, acc[0][0]);
    acc[0][1] = bwmma(ad0, bh1, acc[0][1]);
    acc[1][0] = bwmma(ad1, bh0, acc[1][0]);
    acc[1][1] = bwmma(ad1, bh1, acc[1][1]);
  }

  // Epilogue: lane = token (one t/p/w lookup per token-tile), elements = d.
  int dOff = khalf * 8;
#pragma unroll
  for (int ti = 0; ti < 2; ++ti) {
    unsigned i = (unsigned)(mb * 128 + (2 * wTg + ti) * 16 + nlo);
    if (i < n_e) {
      int t, p; float w;
      if (e == NEXP) { t = (int)i; p = 2; w = 1.0f; }
      else {
        t = ltok[e * T_TOK + i];
        p = lk[e * T_TOK + i];
        w = tokw[2 * t + p];
      }
      size_t base = ((size_t)p * T_TOK + t) * DMODEL + db * 128;
#pragma unroll
      for (int di = 0; di < 2; ++di) {
        size_t cb = base + (2 * wDg + di) * 16 + dOff;
#pragma unroll
        for (int el = 0; el < 8; el += 2) {
          float2 s;
          s.x = w * acc[di][ti][el];
          s.y = w * acc[di][ti][el + 1];
          *(float2*)(&contrib[cb + el]) = s;
        }
      }
    }
  }
}

// ---------------- phase 5: combine three planes -> out ----------------
__global__ __launch_bounds__(256) void combine_kernel(const float* __restrict__ contrib,
                                                      float* __restrict__ out) {
  size_t i = ((size_t)blockIdx.x * 256 + threadIdx.x) * 4;
  const size_t plane = (size_t)T_TOK * DMODEL;
  float4 a = *(const float4*)(contrib + i);
  float4 b = *(const float4*)(contrib + plane + i);
  float4 c = *(const float4*)(contrib + 2 * plane + i);
  float4 o;
  o.x = a.x + b.x + c.x; o.y = a.y + b.y + c.y;
  o.z = a.z + b.z + c.z; o.w = a.w + b.w + c.w;
  *(float4*)(out + i) = o;
}

// ---------------- launcher ----------------
extern "C" void kernel_launch(void* const* d_in, const int* in_sizes, int n_in,
                              void* d_out, int out_size, void* d_ws, size_t ws_size,
                              hipStream_t stream) {
  const float* x   = (const float*)d_in[0];
  const float* rw  = (const float*)d_in[1];
  const float* eb  = (const float*)d_in[2];
  const float* gw  = (const float*)d_in[3];
  const float* uw  = (const float*)d_in[4];
  const float* dw  = (const float*)d_in[5];
  const float* sgw = (const float*)d_in[6];
  const float* suw = (const float*)d_in[7];
  const float* sdw = (const float*)d_in[8];
  float* out = (float*)d_out;

  char* ws = (char*)d_ws;
  bf16*     xbf     = (bf16*)(ws + OFF_XBF);
  bf16*     hidden  = (bf16*)(ws + OFF_HID);
  float*    contrib = (float*)(ws + OFF_CTR);
  float*    tokw    = (float*)(ws + OFF_TW);
  int*      ltok    = (int*)(ws + OFF_LT);
  int*      lkk     = (int*)(ws + OFF_LK);
  unsigned* counts  = (unsigned*)(ws + OFF_CNT);
  unsigned* offs    = (unsigned*)(ws + OFF_OFS);

  hipMemsetAsync(counts, 0, 64, stream);
  convert_x_kernel<<<(T_TOK * DMODEL) / 8 / 256, 256, 0, stream>>>(x, xbf);
  router_kernel<<<T_TOK / 8, 256, 0, stream>>>(x, rw, eb, tokw, ltok, lkk, counts);
  scan_kernel<<<1, 32, 0, stream>>>(counts, offs);
  gateup_kernel<<<dim3(64, HIDDEN / 128, NEXP + 1), 512, 0, stream>>>(
      xbf, gw, uw, sgw, suw, ltok, counts, offs, hidden);
  down_kernel<<<dim3(64, DMODEL / 128, NEXP + 1), 512, 0, stream>>>(
      hidden, dw, sdw, ltok, lkk, tokw, counts, offs, contrib);
  combine_kernel<<<(T_TOK * DMODEL) / 4 / 256, 256, 0, stream>>>(contrib, out);
}
